// Compression_73100343378148
// MI455X (gfx1250) — matrix operand order repaired and checked by reading, not compiled, not run on hardware
//
#include <hip/hip_runtime.h>

#define N_ 32768
#define D_ 1024
#define C_ 256
#define GAMMA_ 1e-4f
#define NCHUNK_ 1024          // split-K chunk over N for the PtX GEMM

typedef __attribute__((ext_vector_type(8)))  unsigned short u16x8;
typedef __attribute__((ext_vector_type(4)))  unsigned short u16x4;
typedef __attribute__((ext_vector_type(16))) __bf16         v16bf;
typedef __attribute__((ext_vector_type(8)))  float          v8f;
typedef __attribute__((ext_vector_type(4)))  float          f32x4;
typedef __attribute__((ext_vector_type(4)))  unsigned int   v4u;
typedef __attribute__((ext_vector_type(4)))  int            v4i;
typedef __attribute__((ext_vector_type(8)))  int            v8i;

#if __has_builtin(__builtin_amdgcn_tensor_load_to_lds)
#define HAVE_TDM 1
#else
#define HAVE_TDM 0
#endif

union BF16x16 { u16x8 h[2]; v16bf v; };

__device__ __forceinline__ unsigned short f2bf(float f) {
    unsigned u = __builtin_bit_cast(unsigned, f);
    u += 0x7FFFu + ((u >> 16) & 1u);          // round-to-nearest-even
    return (unsigned short)(u >> 16);
}

// A-matrix (16x32 bf16, MxK): lane half 'hf', VGPR v: K = (v<4?0:16) + hf*8 + (v%4)*2
__device__ __forceinline__ v16bf load_a16(const unsigned short* rowk0, int hf) {
    BF16x16 u;
    u.h[0] = *(const u16x8*)(rowk0 + hf * 8);
    u.h[1] = *(const u16x8*)(rowk0 + 16 + hf * 8);
    return u.v;
}
// B-matrix (32x16 bf16, KxN): lane half 'hf', VGPR v: K = (v<4?0:16) + hf*8 + (v%4)*2
__device__ __forceinline__ v16bf load_b16(const unsigned short* colk0, int hf) {
    BF16x16 u;
    u.h[0] = *(const u16x8*)(colk0 + hf * 8);
    u.h[1] = *(const u16x8*)(colk0 + 16 + hf * 8);
    return u.v;
}

#if HAVE_TDM
// Issue a TDM 2D tile load: 32 rows x 32 elems (bf16), row stride N_ elements,
// from Pt (row-major [C,N]) into LDS at lds_off.  D# per CDNA5 ISA.
__device__ __forceinline__ void tdm_load_a(const unsigned short* gsrc, unsigned lds_off) {
    unsigned long long ga = (unsigned long long)gsrc;
    v4u g0;
    g0[0] = 1u;                                            // count=1, user D#
    g0[1] = lds_off;                                       // lds_addr (bytes)
    g0[2] = (unsigned)(ga & 0xFFFFFFFFull);                // global_addr lo
    g0[3] = (unsigned)((ga >> 32) & 0x01FFFFFFull) | (2u << 30); // addr hi | type=2
    v8i g1;
    g1[0] = 0x00010000;                                    // data_size=1 (2 bytes)
    g1[1] = (int)(((unsigned)N_ & 0xFFFFu) << 16);         // tensor_dim0 lo16
    g1[2] = (int)((((unsigned)N_ >> 16) & 0xFFFFu) | ((unsigned)C_ << 16)); // td0 hi | td1 lo
    g1[3] = (int)(32u << 16);                              // td1 hi | tile_dim0=32
    g1[4] = 32;                                            // tile_dim1=32, tile_dim2=0
    g1[5] = (int)N_;                                       // tensor_dim0_stride lo32
    g1[6] = 0;                                             // stride hi | td1_stride lo
    g1[7] = 0;
    v4i gz = {0, 0, 0, 0};
#if defined(__clang_major__) && __clang_major__ >= 23
    v8i gz8 = {0, 0, 0, 0, 0, 0, 0, 0};
    __builtin_amdgcn_tensor_load_to_lds(g0, g1, gz, gz, gz8, 0);
#else
    __builtin_amdgcn_tensor_load_to_lds(g0, g1, gz, gz, 0);
#endif
}
#endif

__global__ __launch_bounds__(256) void k_cvt_bf16x4(const float* __restrict__ src,
                                                    unsigned short* __restrict__ dst, int n4) {
    int i = blockIdx.x * blockDim.x + threadIdx.x;
    if (i >= n4) return;
    f32x4 v = ((const f32x4*)src)[i];
    u16x4 o;
    o.x = f2bf(v.x); o.y = f2bf(v.y); o.z = f2bf(v.z); o.w = f2bf(v.w);
    ((u16x4*)dst)[i] = o;
}

__global__ __launch_bounds__(256) void k_zero_f32(float* __restrict__ p, int n) {
    int i = blockIdx.x * blockDim.x + threadIdx.x;
    if (i < n) p[i] = 0.0f;
}

// ---------------------------------------------------------------------------
// Kernel 2: logits = X @ W^T + b, P = softmax(logits).  Block = 32 rows x 256 cols.
// ---------------------------------------------------------------------------
__global__ __launch_bounds__(256) void k_logits_softmax(
    const unsigned short* __restrict__ Xb,   // [N,D] bf16
    const unsigned short* __restrict__ Wb,   // [C,D] bf16
    const float* __restrict__ bias,          // [C]
    unsigned short* __restrict__ P,          // [N,C] bf16
    unsigned short* __restrict__ Pt)         // [C,N] bf16
{
    __shared__ float s_log[32 * 256];
    __shared__ float s_red[32 * 8];
    __shared__ float s_inv[32];

    const int tid  = threadIdx.x;
    const int w    = tid >> 5, lane = tid & 31;
    const int hm   = lane & 15, hf = lane >> 4;
    const int rowGroup = w >> 2, colWave = w & 3;
    const int row0 = blockIdx.x * 32 + rowGroup * 16;
    const int col0 = colWave * 64;

    v8f acc[4] = {};
    const unsigned short* Arow = Xb + (size_t)(row0 + hm) * D_;
    for (int k0 = 0; k0 < D_; k0 += 32) {
        __builtin_prefetch(Arow + k0 + 128, 0, 0);
        v16bf a = load_a16(Arow + k0, hf);
        v16bf bt[4];
#pragma unroll
        for (int t = 0; t < 4; ++t)
            bt[t] = load_b16(Wb + (size_t)(col0 + t * 16 + hm) * D_ + k0, hf);
#pragma unroll
        for (int t = 0; t < 4; ++t)
            acc[t] = __builtin_amdgcn_wmma_f32_16x16x32_bf16(
                false, a, false, bt[t], (short)0, acc[t], false, false);
    }
    // dump logits (+bias) to LDS.  C/D layout: row = v + 8*hf, col = hm.
#pragma unroll
    for (int t = 0; t < 4; ++t) {
        int col = col0 + t * 16 + hm;
        float bv = bias[col];
#pragma unroll
        for (int v = 0; v < 8; ++v) {
            int lr = rowGroup * 16 + v + 8 * hf;
            s_log[lr * 256 + col] = acc[t][v] + bv;
        }
    }
    __syncthreads();

    // softmax over C=256 per row; 8 threads/row, 32 cols each
    const int r = tid & 31, seg = tid >> 5;
    const int base = r * 256 + seg * 32;
    float m = -3.4e38f;
#pragma unroll
    for (int j = 0; j < 32; ++j) m = fmaxf(m, s_log[base + j]);
    s_red[r * 8 + seg] = m;
    __syncthreads();
    float rmax = s_red[r * 8];
#pragma unroll
    for (int j = 1; j < 8; ++j) rmax = fmaxf(rmax, s_red[r * 8 + j]);
    __syncthreads();
    float s = 0.0f;
#pragma unroll
    for (int j = 0; j < 32; ++j) {
        float e = __expf(s_log[base + j] - rmax);
        s_log[base + j] = e;
        s += e;
    }
    s_red[r * 8 + seg] = s;
    __syncthreads();
    if (tid < 32) {
        float t = 0.0f;
#pragma unroll
        for (int j = 0; j < 8; ++j) t += s_red[tid * 8 + j];
        s_inv[tid] = 1.0f / t;
    }
    __syncthreads();

    // write P [N,C] — 32 contiguous cols of one row, 4x 16B stores
    {
        float inv = s_inv[r];
        size_t prow = (size_t)(blockIdx.x * 32 + r) * C_ + seg * 32;
#pragma unroll
        for (int i8 = 0; i8 < 4; ++i8) {
            u16x8 pk;
#pragma unroll
            for (int j = 0; j < 8; ++j) pk[j] = f2bf(s_log[base + i8 * 8 + j] * inv);
            *(u16x8*)(P + prow + i8 * 8) = pk;
        }
    }
    // write Pt [C,N] — thread owns column c=tid, 32 contiguous n's, 4x 16B stores
    {
        int c = tid;
        size_t pbase = (size_t)c * N_ + (size_t)blockIdx.x * 32;
#pragma unroll
        for (int i8 = 0; i8 < 4; ++i8) {
            u16x8 pk;
#pragma unroll
            for (int j = 0; j < 8; ++j) {
                int rr = i8 * 8 + j;
                pk[j] = f2bf(s_log[rr * 256 + c] * s_inv[rr]);
            }
            *(u16x8*)(Pt + pbase + i8 * 8) = pk;
        }
    }
}

// ---------------------------------------------------------------------------
// Kernel 3: PtXt[d][c] += sum_{n in chunk} P[n][c] * X[n][d]   (= (P^T X)^T)
// Block tile: 32 c x 256 d; wave = 2 c-tiles x 2 d-tiles (4 WMMA / stage).
// X staged transposed into double-buffered LDS (1 barrier per 32-n stage);
// the Pt A-slab (32c x 32n) is loaded by the Tensor Data Mover when available.
// ---------------------------------------------------------------------------
__global__ __launch_bounds__(256) void k_ptx(
    const unsigned short* __restrict__ Xb,   // [N,D] bf16
    const unsigned short* __restrict__ Pt,   // [C,N] bf16
    float* __restrict__ PtXt)                // [D,C] f32 accumulator
{
    __shared__ unsigned short sXT[2][256 * 40];  // transposed X: [d_local][n+pad]
#if HAVE_TDM
    __shared__ unsigned short sA[2][32 * 32];    // Pt slab: [c_local][n_local]
#endif

    const int tid = threadIdx.x;
    const int w   = tid >> 5, lane = tid & 31;
    const int hm  = lane & 15, hf = lane >> 4;
    const int d0  = blockIdx.x * 256;
    const int c0  = blockIdx.y * 32;
    const int nbase = blockIdx.z * NCHUNK_;
    const int NS = NCHUNK_ / 32;

    v8f acc[2][2] = {};
#if !HAVE_TDM
    const unsigned short* Arow0 = Pt + (size_t)(c0 + hm) * N_;
    const unsigned short* Arow1 = Pt + (size_t)(c0 + 16 + hm) * N_;
#endif

    // ---- stage 0 prologue ----
#pragma unroll
    for (int i = 0; i < 4; ++i) {                  // 1024 16B vectors / 256 thr
        int vsel = tid * 4 + i;
        int row  = vsel >> 5;                      // n_local 0..31
        int seg  = vsel & 31;                      // d chunk of 8
        u16x8 g = *(const u16x8*)(Xb + (size_t)(nbase + row) * D_ + d0 + seg * 8);
#pragma unroll
        for (int j = 0; j < 8; ++j) sXT[0][(seg * 8 + j) * 40 + row] = g[j];
    }
#if HAVE_TDM
    if (w == 0) {
        tdm_load_a(Pt + (size_t)c0 * N_ + nbase, (unsigned)(size_t)&sA[0][0]);
        __builtin_amdgcn_s_wait_tensorcnt(0);
    }
#endif
    __syncthreads();

    for (int s = 0; s < NS; ++s) {
        const int cur = s & 1, nxt = cur ^ 1;
        const int n0 = nbase + s * 32;
        const bool more = (s + 1 < NS);

#if HAVE_TDM
        if (more && w == 0)                         // async: overlaps stage+compute
            tdm_load_a(Pt + (size_t)c0 * N_ + n0 + 32, (unsigned)(size_t)&sA[nxt][0]);
#endif
        if (more) {
#pragma unroll
            for (int i = 0; i < 4; ++i) {
                int vsel = tid * 4 + i;
                int row  = vsel >> 5;
                int seg  = vsel & 31;
                u16x8 g = *(const u16x8*)(Xb + (size_t)(n0 + 32 + row) * D_ + d0 + seg * 8);
#pragma unroll
                for (int j = 0; j < 8; ++j) sXT[nxt][(seg * 8 + j) * 40 + row] = g[j];
            }
        }

        // ---- compute current stage: 2 c-tiles x 2 d-tiles ----
        v16bf a0, a1;
#if HAVE_TDM
        a0 = load_a16(&sA[cur][hm * 32], hf);
        a1 = load_a16(&sA[cur][(16 + hm) * 32], hf);
#else
        a0 = load_a16(Arow0 + n0, hf);
        a1 = load_a16(Arow1 + n0, hf);
#endif
        v16bf b0 = load_b16(&sXT[cur][(w * 32 + hm) * 40], hf);
        v16bf b1 = load_b16(&sXT[cur][(w * 32 + 16 + hm) * 40], hf);
        acc[0][0] = __builtin_amdgcn_wmma_f32_16x16x32_bf16(false, a0, false, b0, (short)0, acc[0][0], false, false);
        acc[0][1] = __builtin_amdgcn_wmma_f32_16x16x32_bf16(false, a0, false, b1, (short)0, acc[0][1], false, false);
        acc[1][0] = __builtin_amdgcn_wmma_f32_16x16x32_bf16(false, a1, false, b0, (short)0, acc[1][0], false, false);
        acc[1][1] = __builtin_amdgcn_wmma_f32_16x16x32_bf16(false, a1, false, b1, (short)0, acc[1][1], false, false);

#if HAVE_TDM
        if (more && w == 0) __builtin_amdgcn_s_wait_tensorcnt(0);
#endif
        __syncthreads();
    }

    // flush: element (m=c, n=d) -> PtXt[d*C + c]
#pragma unroll
    for (int ci = 0; ci < 2; ++ci) {
#pragma unroll
        for (int di = 0; di < 2; ++di) {
            int d = d0 + w * 32 + di * 16 + hm;
#pragma unroll
            for (int v = 0; v < 8; ++v) {
                int c = c0 + ci * 16 + v + 8 * hf;
                unsafeAtomicAdd(&PtXt[(size_t)d * C_ + c], acc[ci][di][v]);
            }
        }
    }
}

// ---------------------------------------------------------------------------
// Kernel 4: out = X + GAMMA * P @ PtX.  Block: 16 rows x 1024 cols, K = C.
// ---------------------------------------------------------------------------
__global__ __launch_bounds__(256) void k_final(
    const float* __restrict__ X,             // [N,D] f32
    const unsigned short* __restrict__ P,    // [N,C] bf16
    const unsigned short* __restrict__ PtXtb,// [D,C] bf16 (transposed PtX)
    float* __restrict__ Out)                 // [N,D] f32
{
    const int tid = threadIdx.x;
    const int w   = tid >> 5, lane = tid & 31;
    const int hm  = lane & 15, hf = lane >> 4;
    const int row0 = blockIdx.x * 16;
    const int col0 = w * 128;

    v8f acc[8] = {};
    const unsigned short* Arow = P + (size_t)(row0 + hm) * C_;
#pragma unroll
    for (int k0 = 0; k0 < C_; k0 += 32) {
        v16bf a = load_a16(Arow + k0, hf);
#pragma unroll
        for (int g = 0; g < 2; ++g) {
            v16bf bt[4];
#pragma unroll
            for (int t = 0; t < 4; ++t)
                bt[t] = load_b16(PtXtb + (size_t)(col0 + (g * 4 + t) * 16 + hm) * C_ + k0, hf);
#pragma unroll
            for (int t = 0; t < 4; ++t)
                acc[g * 4 + t] = __builtin_amdgcn_wmma_f32_16x16x32_bf16(
                    false, a, false, bt[t], (short)0, acc[g * 4 + t], false, false);
        }
    }
#pragma unroll
    for (int t = 0; t < 8; ++t) {
        int col = col0 + t * 16 + hm;
#pragma unroll
        for (int v = 0; v < 8; ++v) {
            int row = row0 + v + 8 * hf;
            size_t idx = (size_t)row * D_ + col;
            Out[idx] = X[idx] + GAMMA_ * acc[t][v];
        }
    }
}

extern "C" void kernel_launch(void* const* d_in, const int* in_sizes, int n_in,
                              void* d_out, int out_size, void* d_ws, size_t ws_size,
                              hipStream_t stream) {
    const float* X  = (const float*)d_in[0];   // [N,D]
    const float* W  = (const float*)d_in[1];   // [C,D]
    const float* b  = (const float*)d_in[2];   // [C]
    float* out      = (float*)d_out;

    char* ws = (char*)d_ws;
    size_t o = 0;
    unsigned short* Xb    = (unsigned short*)(ws + o); o += (size_t)N_ * D_ * 2; // 64 MiB
    unsigned short* Wb    = (unsigned short*)(ws + o); o += (size_t)C_ * D_ * 2; // 0.5 MiB
    unsigned short* P     = (unsigned short*)(ws + o); o += (size_t)N_ * C_ * 2; // 16 MiB
    unsigned short* Pt    = (unsigned short*)(ws + o); o += (size_t)C_ * N_ * 2; // 16 MiB
    float*          PtXt  = (float*)(ws + o);          o += (size_t)D_ * C_ * 4; // 1 MiB
    unsigned short* PtXtb = (unsigned short*)(ws + o);                           // 0.5 MiB

    // 1) convert X, W to bf16; zero the PtX accumulator
    {
        int n4 = N_ * D_ / 4;
        k_cvt_bf16x4<<<(n4 + 255) / 256, 256, 0, stream>>>(X, Xb, n4);
    }
    {
        int n4 = C_ * D_ / 4;
        k_cvt_bf16x4<<<(n4 + 255) / 256, 256, 0, stream>>>(W, Wb, n4);
    }
    k_zero_f32<<<(D_ * C_ + 255) / 256, 256, 0, stream>>>(PtXt, D_ * C_);

    // 2) P = softmax(X @ W^T + b); also emit P^T
    k_logits_softmax<<<N_ / 32, 256, 0, stream>>>(Xb, Wb, b, P, Pt);

    // 3) PtXt = (P^T X)^T with split-K atomics over N
    k_ptx<<<dim3(D_ / 256, C_ / 32, N_ / NCHUNK_), 256, 0, stream>>>(Xb, Pt, PtXt);

    // 4) PtXt -> bf16
    {
        int n4 = D_ * C_ / 4;
        k_cvt_bf16x4<<<(n4 + 255) / 256, 256, 0, stream>>>(PtXt, PtXtb, n4);
    }

    // 5) out = X + GAMMA * P @ PtX
    k_final<<<N_ / 16, 256, 0, stream>>>(X, P, PtXtb, out);
}
